// PFGRUCell_26242250179158
// MI455X (gfx1250) — matrix operand order repaired and checked by reading, not compiled, not run on hardware
//
#include <hip/hip_runtime.h>
#include <hip/hip_bf16.h>

typedef __attribute__((ext_vector_type(16))) _Float16 v16h;
typedef __attribute__((ext_vector_type(8)))  _Float16 v8h;
typedef __attribute__((ext_vector_type(8)))  float    v8f;

#define NPART 65536
#define HDIM  128

// ---- workspace layout (bytes) ----
#define OFF_WZ    0                       // 64 KB packed f16 Wz fragments
#define OFF_WR    (64*1024)               // 64 KB
#define OFF_WN    (128*1024)              // 128 KB
#define OFF_SCAL  (256*1024)              // scalars
#define OFF_PARTA (257*1024)              // 256 floats
#define OFF_PARTB (258*1024)
#define OFF_BSUM  (259*1024)
#define OFF_BOFF  (260*1024)
#define OFF_MEAN  (261*1024)
#define OFF_IDX   (262*1024)              // N ints
#define OFF_LRAW  (OFF_IDX   + 256*1024)  // N floats
#define OFF_X     (OFF_LRAW  + 256*1024)
#define OFF_P1PRE (OFF_X     + 256*1024)
#define OFF_CDF   (OFF_P1PRE + 256*1024)
#define OFF_P1N   (OFF_CDF   + 256*1024)
#define OFF_H1    (OFF_P1N   + 256*1024)  // N*128 floats (32 MB)

// scalar slots
#define SL_OBSC 0
#define SL_GMAX1 1
#define SL_LSE1 2
#define SL_TOTAL 3
#define SL_GMAX2 4
#define SL_LSE2 5

__device__ __forceinline__ float sigmoidf_(float x) { return 1.f / (1.f + __expf(-x)); }

// ---------------------------------------------------------------------------
// K0: pack f32 weights into f16 WMMA B-fragment order:
//   flat half index = ((nt*8 + kt)*32 + lane)*16 + j
//   lane: n = nt*16 + (lane&15), g = lane>>4 ;  K = kt*32 + (j>>3)*16 + g*8 + (j&7)
// Also: obs_const = dot(input, Wobs[128:256]) + bobs
// ---------------------------------------------------------------------------
__global__ __launch_bounds__(256) void k_pack(
    const float* __restrict__ Wz, const float* __restrict__ Wr,
    const float* __restrict__ Wn, const float* __restrict__ input_,
    const float* __restrict__ Wobs, const float* __restrict__ bobs,
    _Float16* __restrict__ pz, _Float16* __restrict__ pr,
    _Float16* __restrict__ pw, float* __restrict__ scal)
{
    int tid  = blockIdx.x * 256 + threadIdx.x;   // 0 .. 131071
    int j    = tid & 15;
    int lane = (tid >> 4) & 31;
    int g    = lane >> 4;
    int nl   = lane & 15;
    if (tid < 32768) {
        int frag = tid >> 9; int kt = frag & 7, nt = frag >> 3;
        pz[tid] = (_Float16)Wz[(nt*16 + nl)*256 + kt*32 + (j>>3)*16 + g*8 + (j&7)];
    } else if (tid < 65536) {
        int t2 = tid - 32768; int frag = t2 >> 9; int kt = frag & 7, nt = frag >> 3;
        pr[t2] = (_Float16)Wr[(nt*16 + nl)*256 + kt*32 + (j>>3)*16 + g*8 + (j&7)];
    } else {
        int t2 = tid - 65536; int frag = t2 >> 9; int kt = frag & 7, nt = frag >> 3; // nt 0..15
        pw[t2] = (_Float16)Wn[(nt*16 + nl)*256 + kt*32 + (j>>3)*16 + g*8 + (j&7)];
    }
    if (tid == 0) {
        float s = bobs[0];
        for (int q = 0; q < 128; ++q) s += input_[q] * Wobs[128 + q];
        scal[SL_OBSC] = s;
    }
}

// ---------------------------------------------------------------------------
// K1: fused GRU GEMMs with WMMA. One wave = 16 particles.
//   z = sigmoid([h0|in] Wz^T + bz)   (8 N-tiles x 8 K-steps)
//   r = sigmoid([h0|in] Wr^T + br);  rh = r*h0 -> LDS (transpose to A layout)
//   n_lin = [rh|in] Wn^T + bn ; n = tanh(mu + eps*softplus(var))
//   h1 = (1-z)*n + z*h0  -> h1_pre ;  lraw = h1 . Wobs[:128]  (shuffle reduce)
// ---------------------------------------------------------------------------
__global__ __launch_bounds__(256) void k_main(
    const float* __restrict__ h0, const float* __restrict__ input_,
    const float* __restrict__ eps,
    const _Float16* __restrict__ WzPk, const _Float16* __restrict__ WrPk,
    const _Float16* __restrict__ WnPk,
    const float* __restrict__ bz, const float* __restrict__ br,
    const float* __restrict__ bn, const float* __restrict__ Wobs,
    float* __restrict__ h1_pre, float* __restrict__ lraw)
{
    __shared__ __align__(16) _Float16 rh[8][16][136];   // per-wave r*h0 tile, padded

    const int lane  = threadIdx.x & 31;
    const int wv    = threadIdx.x >> 5;
    const int m0    = (blockIdx.x * 8 + wv) * 16;
    const int mlane = lane & 15;
    const int g     = lane >> 4;

    // ---- build A fragments of [h0 | input] : 8 K-tiles of 16x32 f16 ----
    v16h Aobs[8];
    {
        const float* hrow = h0 + (size_t)(m0 + mlane) * HDIM;
        float t0[8] __attribute__((aligned(16)));
        float t1[8] __attribute__((aligned(16)));
#pragma unroll
        for (int kt = 0; kt < 4; ++kt) {
            int b0 = kt*32 + g*8;
            *(float4*)&t0[0] = *(const float4*)(hrow + b0);
            *(float4*)&t0[4] = *(const float4*)(hrow + b0 + 4);
            *(float4*)&t1[0] = *(const float4*)(hrow + b0 + 16);
            *(float4*)&t1[4] = *(const float4*)(hrow + b0 + 20);
            v16h a;
#pragma unroll
            for (int q = 0; q < 8; ++q) { a[q] = (_Float16)t0[q]; a[8+q] = (_Float16)t1[q]; }
            Aobs[kt] = a;
        }
#pragma unroll
        for (int kt = 4; kt < 8; ++kt) {
            int b0 = kt*32 - 128 + g*8;
            *(float4*)&t0[0] = *(const float4*)(input_ + b0);
            *(float4*)&t0[4] = *(const float4*)(input_ + b0 + 4);
            *(float4*)&t1[0] = *(const float4*)(input_ + b0 + 16);
            *(float4*)&t1[4] = *(const float4*)(input_ + b0 + 20);
            v16h a;
#pragma unroll
            for (int q = 0; q < 8; ++q) { a[q] = (_Float16)t0[q]; a[8+q] = (_Float16)t1[q]; }
            Aobs[kt] = a;
        }
    }

    const v16h* Bz = (const v16h*)WzPk;
    const v16h* Br = (const v16h*)WrPk;
    const v16h* Bn = (const v16h*)WnPk;

    // ---- z gate: keep all 8 tiles resident ----
    v8f zt[8];
#pragma unroll
    for (int t = 0; t < 8; ++t) {
        v8f acc = {};
#pragma unroll
        for (int kt = 0; kt < 8; ++kt)
            acc = __builtin_amdgcn_wmma_f32_16x16x32_f16(false, Aobs[kt], false,
                    Bz[((size_t)(t*8 + kt))*32 + lane], (short)0, acc, false, false);
        float bias = bz[t*16 + mlane];
#pragma unroll
        for (int q = 0; q < 8; ++q) acc[q] = sigmoidf_(acc[q] + bias);
        zt[t] = acc;
    }

    // ---- r gate -> rh = r*h0 into LDS (transpose C-layout -> A-layout) ----
#pragma unroll
    for (int t = 0; t < 8; ++t) {
        v8f acc = {};
#pragma unroll
        for (int kt = 0; kt < 8; ++kt)
            acc = __builtin_amdgcn_wmma_f32_16x16x32_f16(false, Aobs[kt], false,
                    Br[((size_t)(t*8 + kt))*32 + lane], (short)0, acc, false, false);
        float bias = br[t*16 + mlane];
        int col = t*16 + mlane;
#pragma unroll
        for (int q = 0; q < 8; ++q) {
            float rv = sigmoidf_(acc[q] + bias);
            float hv = h0[(size_t)(m0 + q + 8*g) * HDIM + col];
            rh[wv][q + 8*g][col] = (_Float16)(rv * hv);
        }
    }
    __syncthreads();

    // ---- A fragments for [rh | input] : rh part from LDS ----
    v16h An[4];
    {
        const _Float16* rrow = &rh[wv][mlane][0];
#pragma unroll
        for (int kt = 0; kt < 4; ++kt) {
            v8h lo = *(const v8h*)(rrow + kt*32 + g*8);
            v8h hi = *(const v8h*)(rrow + kt*32 + 16 + g*8);
            v16h a;
#pragma unroll
            for (int q = 0; q < 8; ++q) { a[q] = lo[q]; a[8+q] = hi[q]; }
            An[kt] = a;
        }
    }

    // ---- n GEMM + fused GRU epilogue + obs GEMV partial ----
    float lacc[8];
#pragma unroll
    for (int q = 0; q < 8; ++q) lacc[q] = 0.f;

#pragma unroll
    for (int t = 0; t < 8; ++t) {
        v8f va = {};   // var tile (cols 128+16t of n_lin)
#pragma unroll
        for (int kt = 0; kt < 4; ++kt)
            va = __builtin_amdgcn_wmma_f32_16x16x32_f16(false, An[kt], false,
                    Bn[((size_t)((8+t)*8 + kt))*32 + lane], (short)0, va, false, false);
#pragma unroll
        for (int kt = 4; kt < 8; ++kt)
            va = __builtin_amdgcn_wmma_f32_16x16x32_f16(false, Aobs[kt], false,
                    Bn[((size_t)((8+t)*8 + kt))*32 + lane], (short)0, va, false, false);
        v8f mu = {};   // mu tile (cols 16t)
#pragma unroll
        for (int kt = 0; kt < 4; ++kt)
            mu = __builtin_amdgcn_wmma_f32_16x16x32_f16(false, An[kt], false,
                    Bn[((size_t)(t*8 + kt))*32 + lane], (short)0, mu, false, false);
#pragma unroll
        for (int kt = 4; kt < 8; ++kt)
            mu = __builtin_amdgcn_wmma_f32_16x16x32_f16(false, Aobs[kt], false,
                    Bn[((size_t)(t*8 + kt))*32 + lane], (short)0, mu, false, false);

        int col = t*16 + mlane;
        float bmu = bn[col], bva = bn[128 + col], wo = Wobs[col];
#pragma unroll
        for (int q = 0; q < 8; ++q) {
            size_t off = (size_t)(m0 + q + 8*g) * HDIM + col;
            float ev = eps[off];
            float hv = h0[off];
            float vv = va[q] + bva;
            float sp = fmaxf(vv, 0.f) + log1pf(__expf(-fabsf(vv)));   // softplus
            float nn = tanhf(mu[q] + bmu + ev * sp);
            float zq = zt[t][q];
            float h1v = (1.f - zq) * nn + zq * hv;
            h1_pre[off] = h1v;
            lacc[q] += h1v * wo;
        }
    }

    // reduce obs partial dot across the 16 lanes of each half-wave
#pragma unroll
    for (int q = 0; q < 8; ++q) {
        float s = lacc[q];
        s += __shfl_xor(s, 1, 32);
        s += __shfl_xor(s, 2, 32);
        s += __shfl_xor(s, 4, 32);
        s += __shfl_xor(s, 8, 32);
        if (mlane == 0) lraw[m0 + q + 8*g] = s;
    }
}

// ---------------------------------------------------------------------------
// Reductions / softmax / scan / resample / gather / mean / MLP
// ---------------------------------------------------------------------------
__global__ __launch_bounds__(256) void k_prex(const float* lraw, const float* p0,
                                              const float* scal, float* x) {
    int i = blockIdx.x * 256 + threadIdx.x;
    x[i] = lraw[i] + scal[SL_OBSC] + p0[i];
}

__global__ __launch_bounds__(256) void k_blockmax(const float* x, float* part) {
    __shared__ float s[256];
    int i = blockIdx.x * 256 + threadIdx.x;
    s[threadIdx.x] = x[i];
    __syncthreads();
    for (int off = 128; off; off >>= 1) {
        if (threadIdx.x < off) s[threadIdx.x] = fmaxf(s[threadIdx.x], s[threadIdx.x + off]);
        __syncthreads();
    }
    if (threadIdx.x == 0) part[blockIdx.x] = s[0];
}

__global__ __launch_bounds__(256) void k_fmax(const float* part, float* scal, int slot) {
    __shared__ float s[256];
    s[threadIdx.x] = part[threadIdx.x];
    __syncthreads();
    for (int off = 128; off; off >>= 1) {
        if (threadIdx.x < off) s[threadIdx.x] = fmaxf(s[threadIdx.x], s[threadIdx.x + off]);
        __syncthreads();
    }
    if (threadIdx.x == 0) scal[slot] = s[0];
}

__global__ __launch_bounds__(256) void k_bsumexp(const float* x, const float* scal,
                                                 int slotG, float* part) {
    __shared__ float s[256];
    float gm = scal[slotG];
    int i = blockIdx.x * 256 + threadIdx.x;
    s[threadIdx.x] = __expf(x[i] - gm);
    __syncthreads();
    for (int off = 128; off; off >>= 1) {
        if (threadIdx.x < off) s[threadIdx.x] += s[threadIdx.x + off];
        __syncthreads();
    }
    if (threadIdx.x == 0) part[blockIdx.x] = s[0];
}

__global__ __launch_bounds__(256) void k_fsumlog(const float* part, float* scal,
                                                 int slotG, int slotOut) {
    __shared__ float s[256];
    s[threadIdx.x] = part[threadIdx.x];
    __syncthreads();
    for (int off = 128; off; off >>= 1) {
        if (threadIdx.x < off) s[threadIdx.x] += s[threadIdx.x + off];
        __syncthreads();
    }
    if (threadIdx.x == 0) scal[slotOut] = scal[slotG] + logf(s[0]);
}

__global__ __launch_bounds__(256) void k_p1pre(const float* x, const float* scal,
                                               float* p1pre, float* cdf) {
    int i = blockIdx.x * 256 + threadIdx.x;
    float p = x[i] - scal[SL_LSE1];
    p1pre[i] = p;
    cdf[i] = 0.5f * __expf(p) + (0.5f / (float)NPART);
}

__global__ __launch_bounds__(256) void k_scanblk(float* cdf, float* bsum) {
    __shared__ float s[256];
    int i = blockIdx.x * 256 + threadIdx.x;
    s[threadIdx.x] = cdf[i];
    __syncthreads();
    for (int off = 1; off < 256; off <<= 1) {
        float t = (threadIdx.x >= off) ? s[threadIdx.x - off] : 0.f;
        __syncthreads();
        s[threadIdx.x] += t;
        __syncthreads();
    }
    cdf[i] = s[threadIdx.x];
    if (threadIdx.x == 255) bsum[blockIdx.x] = s[255];
}

__global__ __launch_bounds__(256) void k_scantop(const float* bsum, float* boff,
                                                 float* scal, int slotTot) {
    __shared__ float s[256];
    float v = bsum[threadIdx.x];
    s[threadIdx.x] = v;
    __syncthreads();
    for (int off = 1; off < 256; off <<= 1) {
        float t = (threadIdx.x >= off) ? s[threadIdx.x - off] : 0.f;
        __syncthreads();
        s[threadIdx.x] += t;
        __syncthreads();
    }
    boff[threadIdx.x] = s[threadIdx.x] - v;                 // exclusive
    if (threadIdx.x == 255) scal[slotTot] = s[255];
}

__global__ __launch_bounds__(256) void k_scanadd(float* cdf, const float* boff) {
    int i = blockIdx.x * 256 + threadIdx.x;
    cdf[i] += boff[blockIdx.x];
}

__global__ __launch_bounds__(256) void k_resample(const float* cdf, const float* p1pre,
                                                  const float* scal, int* idx, float* p1n) {
    int i = blockIdx.x * 256 + threadIdx.x;
    float total = scal[SL_TOTAL];
    float u = ((float)i + 0.5f) * (total / (float)NPART);   // systematic resampling
    int j = 0;
    for (int step = 32768; step > 0; step >>= 1) {
        int t = j + step;
        if (t < NPART && cdf[t - 1] < u) j = t;
    }
    idx[i] = j;
    float pn = __expf(p1pre[j]);
    pn = pn / (0.5f * pn + 0.5f / (float)NPART);
    p1n[i] = logf(pn);
}

__global__ __launch_bounds__(256) void k_p1fin(const float* p1n, const float* scal,
                                               float* p1out) {
    int i = blockIdx.x * 256 + threadIdx.x;
    p1out[i] = p1n[i] - scal[SL_LSE2];
}

__global__ __launch_bounds__(256) void k_gather(const float4* __restrict__ h1p,
                                                const int* __restrict__ idx,
                                                float4* __restrict__ h1o) {
    int gid = blockIdx.x * 256 + threadIdx.x;   // N*32 threads
    int i = gid >> 5, c = gid & 31;
    h1o[(size_t)i * 32 + c] = h1p[(size_t)idx[i] * 32 + c];
}

__global__ void k_zero(float* mean) { mean[threadIdx.x] = 0.f; }

__global__ __launch_bounds__(128) void k_mean(const float* __restrict__ p1out,
                                              const float* __restrict__ h1o,
                                              float* __restrict__ mean) {
    int f = threadIdx.x;
    float acc = 0.f;
    for (int i = blockIdx.x; i < NPART; i += 256)
        acc += __expf(p1out[i]) * h1o[(size_t)i * HDIM + f];
    atomicAdd(&mean[f], acc);
}

__global__ void k_mlp(const float* mean, const float* W1, const float* b1,
                      const float* W2, const float* b2, float* out) {
    __shared__ float t1[24];
    int lane = threadIdx.x;
    if (lane < 24) {
        float s = b1[lane];
        for (int q = 0; q < 128; ++q) s += W1[lane * 128 + q] * mean[q];
        t1[lane] = fmaxf(s, 0.f);
    }
    __syncthreads();
    if (lane < 2) {
        float s = b2[lane];
        for (int q = 0; q < 24; ++q) s += W2[lane * 24 + q] * t1[q];
        out[lane] = s;
    }
}

// ---------------------------------------------------------------------------
extern "C" void kernel_launch(void* const* d_in, const int* in_sizes, int n_in,
                              void* d_out, int out_size, void* d_ws, size_t ws_size,
                              hipStream_t stream) {
    const float* input_ = (const float*)d_in[0];
    const float* h0   = (const float*)d_in[1];
    const float* p0   = (const float*)d_in[2];
    const float* eps  = (const float*)d_in[3];
    const float* Wz   = (const float*)d_in[4];
    const float* bz   = (const float*)d_in[5];
    const float* Wr   = (const float*)d_in[6];
    const float* br   = (const float*)d_in[7];
    const float* Wn   = (const float*)d_in[8];
    const float* bn   = (const float*)d_in[9];
    const float* Wobs = (const float*)d_in[10];
    const float* bobs = (const float*)d_in[11];
    const float* W1   = (const float*)d_in[12];
    const float* b1   = (const float*)d_in[13];
    const float* W2   = (const float*)d_in[14];
    const float* b2   = (const float*)d_in[15];

    char* ws = (char*)d_ws;
    _Float16* pz   = (_Float16*)(ws + OFF_WZ);
    _Float16* pr   = (_Float16*)(ws + OFF_WR);
    _Float16* pw   = (_Float16*)(ws + OFF_WN);
    float* scal    = (float*)(ws + OFF_SCAL);
    float* partA   = (float*)(ws + OFF_PARTA);
    float* partB   = (float*)(ws + OFF_PARTB);
    float* bsum    = (float*)(ws + OFF_BSUM);
    float* boff    = (float*)(ws + OFF_BOFF);
    float* mean    = (float*)(ws + OFF_MEAN);
    int*   idx     = (int*)  (ws + OFF_IDX);
    float* lraw    = (float*)(ws + OFF_LRAW);
    float* xarr    = (float*)(ws + OFF_X);
    float* p1pre   = (float*)(ws + OFF_P1PRE);
    float* cdf     = (float*)(ws + OFF_CDF);
    float* p1n     = (float*)(ws + OFF_P1N);
    float* h1pre   = (float*)(ws + OFF_H1);

    float* out    = (float*)d_out;
    float* locOut = out;                                   // 2 floats
    float* h1Out  = out + 2;                               // N*128
    float* p1Out  = out + 2 + (size_t)NPART * HDIM;        // N

    k_pack<<<512, 256, 0, stream>>>(Wz, Wr, Wn, input_, Wobs, bobs, pz, pr, pw, scal);
    k_main<<<512, 256, 0, stream>>>(h0, input_, eps, pz, pr, pw, bz, br, bn, Wobs, h1pre, lraw);

    k_prex<<<256, 256, 0, stream>>>(lraw, p0, scal, xarr);
    k_blockmax<<<256, 256, 0, stream>>>(xarr, partA);
    k_fmax<<<1, 256, 0, stream>>>(partA, scal, SL_GMAX1);
    k_bsumexp<<<256, 256, 0, stream>>>(xarr, scal, SL_GMAX1, partB);
    k_fsumlog<<<1, 256, 0, stream>>>(partB, scal, SL_GMAX1, SL_LSE1);
    k_p1pre<<<256, 256, 0, stream>>>(xarr, scal, p1pre, cdf);

    k_scanblk<<<256, 256, 0, stream>>>(cdf, bsum);
    k_scantop<<<1, 256, 0, stream>>>(bsum, boff, scal, SL_TOTAL);
    k_scanadd<<<256, 256, 0, stream>>>(cdf, boff);
    k_resample<<<256, 256, 0, stream>>>(cdf, p1pre, scal, idx, p1n);

    k_blockmax<<<256, 256, 0, stream>>>(p1n, partA);
    k_fmax<<<1, 256, 0, stream>>>(partA, scal, SL_GMAX2);
    k_bsumexp<<<256, 256, 0, stream>>>(p1n, scal, SL_GMAX2, partB);
    k_fsumlog<<<1, 256, 0, stream>>>(partB, scal, SL_GMAX2, SL_LSE2);
    k_p1fin<<<256, 256, 0, stream>>>(p1n, scal, p1Out);

    k_gather<<<8192, 256, 0, stream>>>((const float4*)h1pre, idx, (float4*)h1Out);
    k_zero<<<1, 128, 0, stream>>>(mean);
    k_mean<<<256, 128, 0, stream>>>(p1Out, h1Out, mean);
    k_mlp<<<1, 32, 0, stream>>>(mean, W1, b1, W2, b2, locOut);
}
